// MultiHeadAttention_85899346358
// MI455X (gfx1250) — matrix operand order repaired and checked by reading, not compiled, not run on hardware
//
#include <hip/hip_runtime.h>

// ---------------- types ----------------
typedef _Float16 v16h __attribute__((ext_vector_type(16)));
typedef _Float16 v8h  __attribute__((ext_vector_type(8)));
typedef _Float16 v4h  __attribute__((ext_vector_type(4)));
typedef float    v8f  __attribute__((ext_vector_type(8)));

// Problem constants (B=4, S=1024, D=1024, H=16, HD=64)
#define BB   4
#define SS   1024
#define DD   1024
#define HH   16
#define HDD  64

// ---------------- WMMA helpers (layouts per CDNA5 ISA) ----------------

// A fragment (16x32 f16): lane holds row M=l16, K = k0 + half*16 + j (contiguous)
__device__ __forceinline__ v16h load_a16(const _Float16* base, int ld, int row0,
                                         int k0, int half_, int l16) {
  const _Float16* p = base + (size_t)(row0 + l16) * ld + k0 + half_ * 16;
  v8h lo = *(const v8h*)p;
  v8h hi = *(const v8h*)(p + 8);
  return __builtin_shufflevector(lo, hi, 0,1,2,3,4,5,6,7,8,9,10,11,12,13,14,15);
}

// B fragment (32x16 f16): lane holds col N=l16, K = k0 + half*16 + j (contiguous 32B).
// `base` is an N-major array with K contiguous per row (i.e. feeds B[k][n] = base[n][k]).
__device__ __forceinline__ v16h load_b16(const _Float16* base, int ld, int col0,
                                         int k0, int half_, int l16) {
  return *(const v16h*)(base + (size_t)(col0 + l16) * ld + k0 + half_ * 16);
}

__device__ __forceinline__ v8f wmma_f16(v16h a, v16h b, v8f c) {
  return __builtin_amdgcn_wmma_f32_16x16x32_f16(false, a, false, b,
                                                (short)0, c, false, false);
}

// ---------------- f32 -> f16 convert (vectorized: float4 -> 4x f16) ----------------
__global__ void cvt_f32_f16(const float* __restrict__ in, _Float16* __restrict__ out, int n4) {
  int i = blockIdx.x * blockDim.x + threadIdx.x;
  if (i < n4) {
    float4 f = ((const float4*)in)[i];
    v4h h;
    h[0] = (_Float16)f.x; h[1] = (_Float16)f.y;
    h[2] = (_Float16)f.z; h[3] = (_Float16)f.w;
    ((v4h*)out)[i] = h;
  }
}

// ---------------- WMMA GEMM: 64x64 per wave (4x4 tiles of 16x16) ----------------
// Computes out[m,n] = sum_k A[m,k]*Bm[n,k] + bias.
// MODE 0: out f16, head layout [B,H,S,HD]; m=token, n=feature; bias[n]
// MODE 1: out f32, flat [4096,1024];        m=token, n=feature; bias[n]
// MODE 2: A = W (rows=neurons), Bm = X (rows=tokens); out f16 transposed [B,H,HD,S]; bias[m]
template <int MODE>
__global__ void __launch_bounds__(32)
gemm64(const _Float16* __restrict__ A, const _Float16* __restrict__ Bm,
       const float* __restrict__ bias, void* __restrict__ outp) {
  const int lane = threadIdx.x & 31;
  const int half_ = lane >> 4;
  const int l16 = lane & 15;
  const int m0 = blockIdx.x * 64;
  const int n0 = blockIdx.y * 64;

  v8f acc[4][4] = {};
  for (int kb = 0; kb < DD; kb += 32) {
    v16h a[4], b[4];
    #pragma unroll
    for (int i = 0; i < 4; i++) a[i] = load_a16(A, DD, m0 + i * 16, kb, half_, l16);
    #pragma unroll
    for (int j = 0; j < 4; j++) b[j] = load_b16(Bm, DD, n0 + j * 16, kb, half_, l16);
    #pragma unroll
    for (int i = 0; i < 4; i++)
      #pragma unroll
      for (int j = 0; j < 4; j++)
        acc[i][j] = wmma_f16(a[i], b[j], acc[i][j]);
  }

  #pragma unroll
  for (int i = 0; i < 4; i++)
    #pragma unroll
    for (int j = 0; j < 4; j++) {
      const int mbase = m0 + i * 16 + half_ * 8;  // + r
      const int n = n0 + j * 16 + l16;
      #pragma unroll
      for (int r = 0; r < 8; r++) {
        const int m = mbase + r;
        float v = acc[i][j][r];
        if (MODE == 0) {
          v += bias[n];
          const int bb = m >> 10, s = m & 1023, h = n >> 6, hd = n & 63;
          ((_Float16*)outp)[(((size_t)(bb * HH + h) * SS + s) * HDD) + hd] = (_Float16)v;
        } else if (MODE == 1) {
          v += bias[n];
          ((float*)outp)[(size_t)m * DD + n] = v;
        } else {  // MODE 2: m = neuron index, n = token index -> Vt[B,H,HD,S]
          v += bias[m];
          const int h = m >> 6, hd = m & 63, bb = n >> 10, s = n & 1023;
          ((_Float16*)outp)[(((size_t)(bb * HH + h) * HDD + hd) * SS) + s] = (_Float16)v;
        }
      }
    }
}

// ---------------- Fused attention: scores -> softmax -> ctx ----------------
// grid (B*H, S/16); block 128 (4 waves); 32KB f16 LDS tile [16][1024].
// Softmax normalization (1/sum) is deferred to the phase-3 accumulator scaling.
__global__ void attn_kernel(const _Float16* __restrict__ Qp, const _Float16* __restrict__ Kp,
                            const _Float16* __restrict__ Vt, const int* __restrict__ mask,
                            _Float16* __restrict__ ctx) {
  __shared__ _Float16 sp[16 * 1024];  // scores, then unnormalized probs (f16), 32 KB
  __shared__ float invLds[16];        // per-query-row 1/sum

  const int bh = blockIdx.x;         // 0..63
  const int b = bh >> 4, h = bh & 15;
  const int q0 = blockIdx.y * 16;    // query block start within S
  const int tid = threadIdx.x;
  const int wave = tid >> 5;
  const int lane = tid & 31;
  const int half_ = lane >> 4;
  const int l16 = lane & 15;

  const _Float16* Qb = Qp + (size_t)bh * SS * HDD;  // [S][HD]
  const _Float16* Kb = Kp + (size_t)bh * SS * HDD;  // [S][HD]
  const _Float16* Vb = Vt + (size_t)bh * HDD * SS;  // [HD][S]

  // ---- Phase 1: scores S[q, key] = (Q . K^T) * 0.125, masked; 64 key-tiles / 4 waves
  // A fragments are loop-invariant (hoisted by compiler).
  for (int ki = wave; ki < 64; ki += 4) {
    v8f acc = {};
    v16h a = load_a16(Qb, HDD, q0, 0, half_, l16);
    v16h bf = load_b16(Kb, HDD, ki * 16, 0, half_, l16);
    acc = wmma_f16(a, bf, acc);
    a = load_a16(Qb, HDD, q0, 32, half_, l16);
    bf = load_b16(Kb, HDD, ki * 16, 32, half_, l16);
    acc = wmma_f16(a, bf, acc);

    const int key = ki * 16 + l16;
    const int mk = mask[b * SS + key];
    #pragma unroll
    for (int r = 0; r < 8; r++) {
      // -30000 fits f16 and exp(-30000 - mx) == 0
      const float sv = mk ? acc[r] * 0.125f : -30000.0f;
      sp[(r + half_ * 8) * 1024 + key] = (_Float16)sv;
    }
  }
  __syncthreads();

  // ---- Phase 2: row max + exp (written back f16 in place) + row sum -> invLds
  // 8 threads per row, 128 cols (16x v8h) each; each thread owns its segment.
  {
    const int row = tid >> 3;
    v8h* rp = (v8h*)(sp + row * 1024 + (tid & 7) * 128);
    float mx = -3.4e38f;
    for (int c = 0; c < 16; c++) {
      v8h hseg = rp[c];
      #pragma unroll
      for (int j = 0; j < 8; j++) mx = fmaxf(mx, (float)hseg[j]);
    }
    for (int s = 1; s < 8; s <<= 1) mx = fmaxf(mx, __shfl_xor(mx, s, 8));
    float sum = 0.f;
    for (int c = 0; c < 16; c++) {
      v8h hseg = rp[c];
      v8h eseg;
      #pragma unroll
      for (int j = 0; j < 8; j++) {
        const float e = __expf((float)hseg[j] - mx);
        sum += e;
        eseg[j] = (_Float16)e;
      }
      rp[c] = eseg;
    }
    for (int s = 1; s < 8; s <<= 1) sum += __shfl_xor(sum, s, 8);
    if ((tid & 7) == 0) invLds[row] = 1.f / sum;
  }
  __syncthreads();

  // ---- Phase 3: ctx[q, d] = (P . V) * inv[q] ; one 16x16 d-tile per wave (HD=64 -> 4)
  {
    const int di = wave;  // 0..3
    v8f acc = {};
    const _Float16* prow = sp + l16 * 1024;  // this lane's query row (A: lane = row)
    for (int kb = 0; kb < SS; kb += 32) {
      const _Float16* p = prow + kb + half_ * 16;
      v8h lo = *(const v8h*)p;         // K = kb + half*16 + 0..7
      v8h hi = *(const v8h*)(p + 8);   // K = kb + half*16 + 8..15
      v16h a = __builtin_shufflevector(lo, hi, 0,1,2,3,4,5,6,7,8,9,10,11,12,13,14,15);
      v16h bf = *(const v16h*)(Vb + (size_t)(di * 16 + l16) * SS + kb + half_ * 16);
      acc = wmma_f16(a, bf, acc);
    }
    #pragma unroll
    for (int r = 0; r < 8; r++) {
      const float inv = invLds[r + half_ * 8];
      const int s = q0 + r + half_ * 8;
      const int n = h * HDD + di * 16 + l16;
      ctx[((size_t)(b * SS + s)) * DD + n] = (_Float16)(acc[r] * inv);
    }
  }
}

// ---------------- host launch ----------------
extern "C" void kernel_launch(void* const* d_in, const int* in_sizes, int n_in,
                              void* d_out, int out_size, void* d_ws, size_t ws_size,
                              hipStream_t stream) {
  (void)in_sizes; (void)n_in; (void)out_size; (void)ws_size;
  const float* q    = (const float*)d_in[0];
  const float* k    = (const float*)d_in[1];
  const float* v    = (const float*)d_in[2];
  const int*   mask = (const int*)d_in[3];
  const float* Wq   = (const float*)d_in[4];
  const float* bq   = (const float*)d_in[5];
  const float* Wk   = (const float*)d_in[6];
  const float* bk   = (const float*)d_in[7];
  const float* Wv   = (const float*)d_in[8];
  const float* bv   = (const float*)d_in[9];
  const float* Wo   = (const float*)d_in[10];
  const float* bo   = (const float*)d_in[11];

  char* ws = (char*)d_ws;
  const size_t MB = 1024 * 1024;
  _Float16* Wq16  = (_Float16*)(ws + 0 * MB);   // 2 MB each
  _Float16* Wk16  = (_Float16*)(ws + 2 * MB);
  _Float16* Wv16  = (_Float16*)(ws + 4 * MB);
  _Float16* Wo16  = (_Float16*)(ws + 6 * MB);
  _Float16* q16   = (_Float16*)(ws + 8 * MB);   // 8 MB each
  _Float16* k16   = (_Float16*)(ws + 16 * MB);
  _Float16* v16_  = (_Float16*)(ws + 24 * MB);
  _Float16* Qp    = (_Float16*)(ws + 32 * MB);  // [B,H,S,HD]
  _Float16* Kp    = (_Float16*)(ws + 40 * MB);  // [B,H,S,HD]
  _Float16* Vt    = (_Float16*)(ws + 48 * MB);  // [B,H,HD,S]
  _Float16* ctx16 = (_Float16*)(ws + 56 * MB);  // [B,S,D]

  const int nW4 = DD * DD / 4;        // 256K float4
  const int nX4 = BB * SS * DD / 4;   // 1M float4
  cvt_f32_f16<<<nW4 / 256, 256, 0, stream>>>(Wq, Wq16, nW4);
  cvt_f32_f16<<<nW4 / 256, 256, 0, stream>>>(Wk, Wk16, nW4);
  cvt_f32_f16<<<nW4 / 256, 256, 0, stream>>>(Wv, Wv16, nW4);
  cvt_f32_f16<<<nW4 / 256, 256, 0, stream>>>(Wo, Wo16, nW4);
  cvt_f32_f16<<<nX4 / 256, 256, 0, stream>>>(q, q16, nX4);
  cvt_f32_f16<<<nX4 / 256, 256, 0, stream>>>(k, k16, nX4);
  cvt_f32_f16<<<nX4 / 256, 256, 0, stream>>>(v, v16_, nX4);

  const dim3 blk(32);
  const dim3 gMN(BB * SS / 64, DD / 64);   // (64, 16): m=token, n=feature
  const dim3 gNM(DD / 64, BB * SS / 64);   // (16, 64): m=neuron, n=token
  gemm64<0><<<gMN, blk, 0, stream>>>(q16, Wq16, bq, Qp);
  gemm64<0><<<gMN, blk, 0, stream>>>(k16, Wk16, bk, Kp);
  gemm64<2><<<gNM, blk, 0, stream>>>(Wv16, v16_, bv, Vt);

  attn_kernel<<<dim3(BB * HH, SS / 16), 128, 0, stream>>>(Qp, Kp, Vt, mask, ctx16);

  gemm64<1><<<gMN, blk, 0, stream>>>(ctx16, Wo16, bo, (float*)d_out);
}
